// DeformableVisTRsegmFinal_34196529610798
// MI455X (gfx1250) — matrix operand order repaired and checked by reading, not compiled, not run on hardware
//
#include <hip/hip_runtime.h>

// ---------------- static problem config (matches reference) ----------------
#define B_T   6
#define NQ_   300
#define DM    256          // d_model
#define NH    8            // heads
#define DH    32           // head dim
#define NL    4            // levels
#define NPT   4            // points
#define DFF_  1024
#define LEN_  20197
#define MROWS (B_T * LEN_)   // 121182 rows of the value projection
#define QROWS (B_T * NQ_)    // 1800 query rows

typedef __attribute__((ext_vector_type(2)))  float   v2f;
typedef __attribute__((ext_vector_type(8)))  float   v8f;
typedef __attribute__((ext_vector_type(16))) __bf16  v16bf;

// ===========================================================================
// Generic fp32 WMMA GEMM: C[M,N] = A[M,K] @ B[K,N] + bias (+resid) (relu)
// One wave computes a 16x64 tile via 4 accumulators of v_wmma_f32_16x16x4_f32.
// flags: bit0 = relu, bit1 = add resid[M,N].
// ===========================================================================
__global__ void gemm_wmma_f32(const float* __restrict__ A,
                              const float* __restrict__ Bm,
                              const float* __restrict__ bias,
                              const float* __restrict__ resid,
                              float* __restrict__ C,
                              int M, int N, int K, int flags)
{
  const int lane  = threadIdx.x & 31;
  const int half  = lane >> 4;
  const int m16   = blockIdx.y;
  const int n64   = blockIdx.x;
  int arow = m16 * 16 + (lane & 15);
  if (arow >= M) arow = M - 1;         // clamp (stores are predicated)
  const int ncol0 = n64 * 64 + (lane & 15);

  v8f acc0 = {}, acc1 = {}, acc2 = {}, acc3 = {};

  for (int k = 0; k < K; k += 4) {
    const int kk = k + half * 2;       // A 16x4 f32 layout: lanes 16-31 hold K=2,3
    v2f a;
    a.x = A[(size_t)arow * K + kk];
    a.y = A[(size_t)arow * K + kk + 1];

    v2f b0, b1, b2, b3;
    b0.x = Bm[(size_t)kk * N + ncol0 +  0]; b0.y = Bm[(size_t)(kk + 1) * N + ncol0 +  0];
    b1.x = Bm[(size_t)kk * N + ncol0 + 16]; b1.y = Bm[(size_t)(kk + 1) * N + ncol0 + 16];
    b2.x = Bm[(size_t)kk * N + ncol0 + 32]; b2.y = Bm[(size_t)(kk + 1) * N + ncol0 + 32];
    b3.x = Bm[(size_t)kk * N + ncol0 + 48]; b3.y = Bm[(size_t)(kk + 1) * N + ncol0 + 48];

    acc0 = __builtin_amdgcn_wmma_f32_16x16x4_f32(false, a, false, b0, (short)0, acc0, false, false);
    acc1 = __builtin_amdgcn_wmma_f32_16x16x4_f32(false, a, false, b1, (short)0, acc1, false, false);
    acc2 = __builtin_amdgcn_wmma_f32_16x16x4_f32(false, a, false, b2, (short)0, acc2, false, false);
    acc3 = __builtin_amdgcn_wmma_f32_16x16x4_f32(false, a, false, b3, (short)0, acc3, false, false);
  }

  v8f accs[4] = {acc0, acc1, acc2, acc3};
#pragma unroll
  for (int j = 0; j < 4; ++j) {
    const int n  = ncol0 + j * 16;
    const float bv = bias ? bias[n] : 0.0f;
#pragma unroll
    for (int v = 0; v < 8; ++v) {
      const int row = m16 * 16 + half * 8 + v;   // C layout: VGPR v -> M=v (+8 hi lanes)
      if (row < M) {
        float val = accs[j][v] + bv;
        if (flags & 1) val = fmaxf(val, 0.0f);
        if (flags & 2) val += resid[(size_t)row * N + n];
        C[(size_t)row * N + n] = val;
      }
    }
  }
}

// ===========================================================================
// Pre-pack W_value (fp32 256x256, row-major) into bf16 WMMA B-fragment order:
// Wpack[((ntile16*8 + kstep)*32 + lane)*16 + e]  == per-lane v16bf fragment.
// B packing here (contiguous k, not AMD table order): lane -> col, VGPR pair v -> K = 16*(lane>=16) + 2v.
// grid = (16 col-tiles, 8 k-steps), block = 32.
// ===========================================================================
__global__ void pack_w_bf16(const float* __restrict__ W, __bf16* __restrict__ Wpack)
{
  const int lane = threadIdx.x & 31;
  const int half = lane >> 4;
  const int nt   = blockIdx.x;            // 16-col tile
  const int ks   = blockIdx.y;            // 32-K step
  const int n    = nt * 16 + (lane & 15);
  v16bf frag;
#pragma unroll
  for (int v = 0; v < 8; ++v) {
    const int kk = ks * 32 + 16 * half + 2 * v;
    frag[2 * v]     = (__bf16)W[(size_t)kk * DM + n];
    frag[2 * v + 1] = (__bf16)W[(size_t)(kk + 1) * DM + n];
  }
  *(v16bf*)(Wpack + ((size_t)(nt * 8 + ks) * 32 + lane) * 16) = frag;
}

// ===========================================================================
// Value projection GEMM: C[M,256] = A[M,256] @ W + bias, bf16 WMMA math with
// fp32 accumulation.  Block = 128 threads (4 waves x 16 rows); the 64-column
// packed-bf16 B tile (32 KB) is staged into LDS once per workgroup with the
// async DMA path (GLOBAL_LOAD_ASYNC_TO_LDS_B128, ASYNCcnt) and then consumed
// by all waves via ds_load_b128.  A fragments stream from HBM as float2 with
// in-register fp32->bf16 conversion (src read once per 64-column block).
// ===========================================================================
__global__ void __launch_bounds__(128)
gemm_value_bf16(const float* __restrict__ A,
                const __bf16* __restrict__ Wpack,
                const float* __restrict__ bias,
                float* __restrict__ C, int M)
{
  constexpr int K = DM, N = DM;
  __shared__ __bf16 ldsB[4 * 8 * 32 * 16];      // 16384 bf16 = 32 KB
  const int t    = threadIdx.x;
  const int lane = t & 31;
  const int w    = t >> 5;
  const int half = lane >> 4;
  const int n64  = blockIdx.x;

  // ---- async stage of this block's packed B tile (32 KB) into LDS ----
  {
    const char* gsrc = (const char*)(Wpack + (size_t)n64 * (4 * 8 * 32 * 16));
    const unsigned ldsBase = (unsigned)(uintptr_t)(&ldsB[0]);
#pragma unroll
    for (int i = 0; i < 16; ++i) {
      const unsigned off = (unsigned)(t * 16 + i * 128 * 16);   // 16B chunks
      asm volatile("global_load_async_to_lds_b128 %0, %1, %2"
                   :: "v"(ldsBase + off), "v"(off), "s"(gsrc)
                   : "memory");
    }
    asm volatile("s_wait_asynccnt 0" ::: "memory");
  }
  __syncthreads();

  int arow = blockIdx.y * 64 + w * 16 + (lane & 15);
  if (arow >= M) arow = M - 1;
  const float* Arow = A + (size_t)arow * K;
  const int ncol0 = n64 * 64 + (lane & 15);

  v8f acc0 = {}, acc1 = {}, acc2 = {}, acc3 = {};

#pragma unroll
  for (int ks = 0; ks < K / 32; ++ks) {
    const int k = ks * 32;
    __builtin_prefetch(Arow + k + 64, 0, 0);
    // A fragment in this code's contiguous k order (matches packed B): pair v -> K = 16*half+2*v
    v16bf a;
#pragma unroll
    for (int v = 0; v < 8; ++v) {
      const int kk = k + half * 16 + 2 * v;
      const float2 f = *(const float2*)(Arow + kk);
      a[2 * v]     = (__bf16)f.x;
      a[2 * v + 1] = (__bf16)f.y;
    }
    // B fragments from LDS: one v16bf (32 B) per 16-col sub-tile
    const v16bf b0 = *(const v16bf*)&ldsB[(((0 * 8) + ks) * 32 + lane) * 16];
    const v16bf b1 = *(const v16bf*)&ldsB[(((1 * 8) + ks) * 32 + lane) * 16];
    const v16bf b2 = *(const v16bf*)&ldsB[(((2 * 8) + ks) * 32 + lane) * 16];
    const v16bf b3 = *(const v16bf*)&ldsB[(((3 * 8) + ks) * 32 + lane) * 16];

    acc0 = __builtin_amdgcn_wmma_f32_16x16x32_bf16(false, a, false, b0, (short)0, acc0, false, false);
    acc1 = __builtin_amdgcn_wmma_f32_16x16x32_bf16(false, a, false, b1, (short)0, acc1, false, false);
    acc2 = __builtin_amdgcn_wmma_f32_16x16x32_bf16(false, a, false, b2, (short)0, acc2, false, false);
    acc3 = __builtin_amdgcn_wmma_f32_16x16x32_bf16(false, a, false, b3, (short)0, acc3, false, false);
  }

  v8f accs[4] = {acc0, acc1, acc2, acc3};
#pragma unroll
  for (int j = 0; j < 4; ++j) {
    const int n  = ncol0 + j * 16;
    const float bv = bias[n];
#pragma unroll
    for (int v = 0; v < 8; ++v) {
      const int row = blockIdx.y * 64 + w * 16 + half * 8 + v;
      if (row < M)
        C[(size_t)row * N + n] = accs[j][v] + bv;
    }
  }
}

// ===========================================================================
// q = query + query_pos
// ===========================================================================
__global__ void add_kernel(const float* __restrict__ a, const float* __restrict__ b,
                           float* __restrict__ o, int n)
{
  const int i = blockIdx.x * blockDim.x + threadIdx.x;
  if (i < n) o[i] = a[i] + b[i];
}

// ===========================================================================
// Per-(b,q,head): softmax over 16 logits + sampling-location computation.
// attn may be in-place (each thread owns its 16 values).
// ===========================================================================
__global__ void softmax_loc_kernel(const float* __restrict__ offr,
                                   const float* __restrict__ attn_in,
                                   const float* __restrict__ refp,
                                   float* __restrict__ attn_out,
                                   float* __restrict__ loc)
{
  static constexpr float LWf[NL] = {152.f, 76.f, 38.f, 19.f};
  static constexpr float LHf[NL] = {100.f, 50.f, 25.f, 13.f};
  const int idx = blockIdx.x * blockDim.x + threadIdx.x;   // over QROWS*NH
  if (idx >= QROWS * NH) return;
  const int bq = idx / NH;
  const int h  = idx - bq * NH;

  const float* ar = attn_in + (size_t)bq * (NH * NL * NPT) + h * (NL * NPT);
  float e[NL * NPT];
  float mx = ar[0];
#pragma unroll
  for (int i = 1; i < NL * NPT; ++i) mx = fmaxf(mx, ar[i]);
  float sum = 0.f;
#pragma unroll
  for (int i = 0; i < NL * NPT; ++i) { e[i] = expf(ar[i] - mx); sum += e[i]; }
  const float inv = 1.f / sum;
  float* ao = attn_out + (size_t)bq * (NH * NL * NPT) + h * (NL * NPT);
#pragma unroll
  for (int i = 0; i < NL * NPT; ++i) ao[i] = e[i] * inv;

  const float* orow = offr + (size_t)bq * (NH * NL * NPT * 2) + h * (NL * NPT * 2);
  const float* rp   = refp + (size_t)bq * (NL * 2);
  float* lp         = loc  + (size_t)idx * (NL * NPT * 2);
#pragma unroll
  for (int l = 0; l < NL; ++l)
#pragma unroll
    for (int p = 0; p < NPT; ++p) {
      const int ip = l * NPT + p;
      lp[ip * 2 + 0] = rp[l * 2 + 0] + orow[ip * 2 + 0] * (1.f / LWf[l]);
      lp[ip * 2 + 1] = rp[l * 2 + 1] + orow[ip * 2 + 1] * (1.f / LHf[l]);
    }
}

// ===========================================================================
// Multi-scale deformable sampling.  Block = (b,q); wave w = head; lane = chan.
// value rows are 128B-coalesced per (head, corner) gather.
// ===========================================================================
__global__ void msdeform_kernel(const float* __restrict__ value,
                                const float* __restrict__ loc,
                                const float* __restrict__ attn,
                                float* __restrict__ msout)
{
  static constexpr int LH[NL] = {100, 50, 25, 13};
  static constexpr int LW[NL] = {152, 76, 38, 19};
  static constexpr int LS[NL] = {0, 15200, 19000, 19950};
  const int bq   = blockIdx.x;
  const int h    = threadIdx.x >> 5;
  const int lane = threadIdx.x & 31;
  const int b    = bq / NQ_;
  const float* locp = loc  + ((size_t)bq * NH + h) * (NL * NPT * 2);
  const float* attp = attn + ((size_t)bq * NH + h) * (NL * NPT);
  float acc = 0.f;
#pragma unroll
  for (int l = 0; l < NL; ++l) {
    const int H = LH[l], W = LW[l];
    const float* vbase = value + ((size_t)b * LEN_ + LS[l]) * DM + h * DH + lane;
    for (int p = 0; p < NPT; ++p) {
      const int ip  = l * NPT + p;
      const float x = locp[ip * 2 + 0] * (float)W - 0.5f;
      const float y = locp[ip * 2 + 1] * (float)H - 0.5f;
      const float xf = floorf(x), yf = floorf(y);
      const float wx = x - xf, wy = y - yf;
      const int x0 = (int)xf, y0 = (int)yf;
      const bool xin0 = (x0 >= 0) && (x0 < W);
      const bool xin1 = (x0 + 1 >= 0) && (x0 + 1 < W);
      const bool yin0 = (y0 >= 0) && (y0 < H);
      const bool yin1 = (y0 + 1 >= 0) && (y0 + 1 < H);
      float s = 0.f;
      if (yin0 && xin0) s += (1.f - wx) * (1.f - wy) * vbase[(size_t)(y0 * W + x0) * DM];
      if (yin0 && xin1) s += wx * (1.f - wy)         * vbase[(size_t)(y0 * W + x0 + 1) * DM];
      if (yin1 && xin0) s += (1.f - wx) * wy         * vbase[(size_t)((y0 + 1) * W + x0) * DM];
      if (yin1 && xin1) s += wx * wy                 * vbase[(size_t)((y0 + 1) * W + x0 + 1) * DM];
      acc += attp[ip] * s;
    }
  }
  msout[(size_t)bq * DM + h * DH + lane] = acc;
}

// ===========================================================================
// LayerNorm over 256 channels: block = row, 256 threads, LDS tree reduce.
// ===========================================================================
__global__ void layernorm_kernel(const float* __restrict__ x, const float* __restrict__ g,
                                 const float* __restrict__ bta, float* __restrict__ y)
{
  __shared__ float red[DM];
  const int row = blockIdx.x, t = threadIdx.x;
  const float v = x[(size_t)row * DM + t];
  red[t] = v;
  __syncthreads();
  for (int s = DM / 2; s > 0; s >>= 1) {
    if (t < s) red[t] += red[t + s];
    __syncthreads();
  }
  const float mean = red[0] * (1.f / DM);
  __syncthreads();
  const float d = v - mean;
  red[t] = d * d;
  __syncthreads();
  for (int s = DM / 2; s > 0; s >>= 1) {
    if (t < s) red[t] += red[t + s];
    __syncthreads();
  }
  const float var = red[0] * (1.f / DM);
  y[(size_t)row * DM + t] = d * rsqrtf(var + 1e-5f) * g[t] + bta[t];
}

// ===========================================================================
extern "C" void kernel_launch(void* const* d_in, const int* in_sizes, int n_in,
                              void* d_out, int out_size, void* d_ws, size_t ws_size,
                              hipStream_t stream)
{
  const float* query     = (const float*)d_in[0];
  const float* query_pos = (const float*)d_in[1];
  const float* refp      = (const float*)d_in[2];
  const float* src       = (const float*)d_in[3];
  const float* W_value   = (const float*)d_in[4];
  const float* b_value   = (const float*)d_in[5];
  const float* W_off     = (const float*)d_in[6];
  const float* b_off     = (const float*)d_in[7];
  const float* W_attn    = (const float*)d_in[8];
  const float* b_attn    = (const float*)d_in[9];
  const float* W_out     = (const float*)d_in[10];
  const float* b_out     = (const float*)d_in[11];
  const float* ln1_g     = (const float*)d_in[12];
  const float* ln1_b     = (const float*)d_in[13];
  const float* W_ffn1    = (const float*)d_in[14];
  const float* b_ffn1    = (const float*)d_in[15];
  const float* W_ffn2    = (const float*)d_in[16];
  const float* b_ffn2    = (const float*)d_in[17];
  const float* ln3_g     = (const float*)d_in[18];
  const float* ln3_b     = (const float*)d_in[19];
  float* out = (float*)d_out;

  // workspace carve-up (floats)
  float* ws = (float*)d_ws;
  size_t o = 0;
  float* value  = ws + o; o += (size_t)MROWS * DM;              // 31.0M
  float* q      = ws + o; o += (size_t)QROWS * DM;
  float* offr   = ws + o; o += (size_t)QROWS * 256;
  float* attn   = ws + o; o += (size_t)QROWS * 128;
  float* loc    = ws + o; o += (size_t)QROWS * NH * NL * NPT * 2;
  float* msout  = ws + o; o += (size_t)QROWS * DM;
  float* t2q    = ws + o; o += (size_t)QROWS * DM;
  float* tgt    = ws + o; o += (size_t)QROWS * DM;
  float* ffn1   = ws + o; o += (size_t)QROWS * DFF_;
  float* ffnsum = ws + o; o += (size_t)QROWS * DM;
  __bf16* Wpack = (__bf16*)(ws + o);                            // 65536 bf16 = 128 KB

  const int qTiles = (QROWS + 15) / 16;   // 113
  const int mBlks  = (MROWS + 63) / 64;   // 1894 (64 rows per 4-wave block)

  // 1) q = query + query_pos
  add_kernel<<<dim3((QROWS * DM + 255) / 256), 256, 0, stream>>>(query, query_pos, q, QROWS * DM);
  // 2) value = src @ W_value + b_value  (pack W once, then bf16 WMMA w/ async LDS)
  pack_w_bf16<<<dim3(16, 8), 32, 0, stream>>>(W_value, Wpack);
  gemm_value_bf16<<<dim3(DM / 64, mBlks), 128, 0, stream>>>(src, Wpack, b_value, value, MROWS);
  // 3) sampling offsets / attention logits (fp32 WMMA)
  gemm_wmma_f32<<<dim3(256 / 64, qTiles), 32, 0, stream>>>(q, W_off, b_off, nullptr, offr,
                                                           QROWS, 256, DM, 0);
  gemm_wmma_f32<<<dim3(128 / 64, qTiles), 32, 0, stream>>>(q, W_attn, b_attn, nullptr, attn,
                                                           QROWS, 128, DM, 0);
  // 4) softmax + sampling locations
  softmax_loc_kernel<<<dim3((QROWS * NH + 255) / 256), 256, 0, stream>>>(offr, attn, refp,
                                                                         attn, loc);
  // 5) deformable bilinear sampling + head-weighted sum
  msdeform_kernel<<<dim3(QROWS), 256, 0, stream>>>(value, loc, attn, msout);
  // 6) output projection + residual -> LN1
  gemm_wmma_f32<<<dim3(DM / 64, qTiles), 32, 0, stream>>>(msout, W_out, b_out, query, t2q,
                                                          QROWS, DM, DM, /*resid*/2);
  layernorm_kernel<<<dim3(QROWS), DM, 0, stream>>>(t2q, ln1_g, ln1_b, tgt);
  // 7) FFN: relu(tgt @ W1 + b1) @ W2 + b2 + tgt -> LN3
  gemm_wmma_f32<<<dim3(DFF_ / 64, qTiles), 32, 0, stream>>>(tgt, W_ffn1, b_ffn1, nullptr, ffn1,
                                                            QROWS, DFF_, DM, /*relu*/1);
  gemm_wmma_f32<<<dim3(DM / 64, qTiles), 32, 0, stream>>>(ffn1, W_ffn2, b_ffn2, tgt, ffnsum,
                                                          QROWS, DM, DFF_, /*resid*/2);
  layernorm_kernel<<<dim3(QROWS), DM, 0, stream>>>(ffnsum, ln3_g, ln3_b, out);
}
